// DWTSelfAttentionND_47433618817130
// MI455X (gfx1250) — compile-verified, matrix operand order checked by reading, not run on hardware
//
#include <hip/hip_runtime.h>
#include <hip/hip_bf16.h>

// ---------------------------------------------------------------------------
// CDNA5 (gfx1250) DWT + axial-MHSA.  Fused per-line attention kernel:
// v_wmma_f32_16x16x32_f16, all operands LDS-resident; every fragment load is
// 2x ds_load_b128, nearly every D-tile store is 1x ds_store_b128.  Kernel is
// templated on line length L so all tile math is compile-time.
// ---------------------------------------------------------------------------

typedef __attribute__((ext_vector_type(16))) _Float16 v16h;
typedef __attribute__((ext_vector_type(8)))  _Float16 v8h;
typedef __attribute__((ext_vector_type(8)))  float    v8f;
typedef __attribute__((ext_vector_type(4)))  float    v4f;

__device__ __forceinline__ v8f wmma16(v16h a, v16h b, v8f c) {
  return __builtin_amdgcn_wmma_f32_16x16x32_f16(
      false, a, false, b, (short)0, c, false, false);
}

// 16x32 operand fragment from a "row-contiguous-K" f16 buffer p[r*ld + k].
// For A: r = M row.  For B: buffer must be N-major (r = N column).
// ISA 7.12.2: lane r(0..15) -> K {0..7,16..23}; lane r+16 -> {8..15,24..31}.
__device__ __forceinline__ v16h frag16(const _Float16* p, int ld, int lane) {
  int r = lane & 15;
  int klo = (lane & 16) ? 8 : 0;
  const _Float16* row = p + r * ld + klo;
  v8h lo = *(const v8h*)(row);
  v8h hi = *(const v8h*)(row + 16);
  return __builtin_shufflevector(lo, hi, 0, 1, 2, 3, 4, 5, 6, 7,
                                 8, 9, 10, 11, 12, 13, 14, 15);
}

// Load 8 consecutive f32 (32B aligned) -> v8f via two b128 LDS loads.
__device__ __forceinline__ v8f load8f(const float* p) {
  v4f a = *(const v4f*)p;
  v4f b = *(const v4f*)(p + 4);
  return __builtin_shufflevector(a, b, 0, 1, 2, 3, 4, 5, 6, 7);
}

// Store f32 16x16 C/D tile to f16 buffer, M-major: p[m*ldc + n]. (8x b16)
__device__ __forceinline__ void store_c_f16(_Float16* p, int ldc, int lane, v8f c) {
  int n = lane & 15;
  int mo = (lane & 16) ? 8 : 0;
#pragma unroll
  for (int r = 0; r < 8; ++r) p[(mo + r) * ldc + n] = (_Float16)c[r];
}

// Store f32 16x16 C/D tile transposed (N-major): p[n*ldT + m].  (1x b128)
__device__ __forceinline__ void store_c_f16T(_Float16* p, int ldT, int lane, v8f c, float mul) {
  int n = lane & 15;
  int mo = (lane & 16) ? 8 : 0;
  v8h v;
#pragma unroll
  for (int r = 0; r < 8; ++r) v[r] = (_Float16)(c[r] * mul);
  *(v8h*)(p + n * ldT + mo) = v;
}

// ---------------------------------------------------------------------------
// Fused axial attention pass.  One workgroup (256 threads / 8 waves) = one
// line of length L (=H if axis==0 else W).  Writes pure attention output.
// ---------------------------------------------------------------------------
template <int L>
__global__ void __launch_bounds__(256) attn_pass_kernel(
    const float* __restrict__ xin, float* __restrict__ xout,
    const float* __restrict__ lng, const float* __restrict__ lnb,
    const float* __restrict__ wq, const float* __restrict__ bq,
    const float* __restrict__ wk, const float* __restrict__ bk,
    const float* __restrict__ wv, const float* __restrict__ bv,
    const float* __restrict__ wp, const float* __restrict__ bp,
    int H, int W, int axis)
{
  constexpr int MT = L >> 4;     // pixel tiles (4 or 8)
  constexpr int KC = L >> 5;     // K-chunks over L (2 or 4)
  extern __shared__ char smem[];

  // f16 region first (16B aligned); all sizes multiples of 16 halves.
  _Float16* sWqT = (_Float16*)smem;        // 128x64  WqT[n][k] = wq[k*128+n]
  _Float16* sWkT = sWqT + 128 * 64;
  _Float16* sWvT = sWkT + 128 * 64;
  _Float16* sWpT = sWvT + 128 * 64;        // 64x128  WpT[n][k] = wp[k*64+n]
  _Float16* sX   = sWpT + 64 * 128;        // L x 64   LN'd input (pixel-major)
  _Float16* sQ   = sX + L * 64;            // L x 128  pixel-major (scale folded)
  _Float16* sK   = sQ + L * 128;           // L x 128  key-major
  _Float16* sVt  = sK + L * 128;           // 128 x L  channel-major
  _Float16* sY   = sVt + 128 * L;          // L x 128  pixel-major
  _Float16* sS   = sY + L * 128;           // L x L    query-major (one head at a time)
  float* sG   = (float*)(sS + L * L);      // 64
  float* sB   = sG + 64;                   // 64
  float* sBq  = sB + 64;                   // 128
  float* sBk  = sBq + 128;
  float* sBv  = sBk + 128;
  float* sBp  = sBv + 128;                 // 64
  float* sRow = sBp + 64;                  // L (reciprocal softmax sums)

  const int tid  = threadIdx.x;
  const int lane = tid & 31;
  const int wid  = __builtin_amdgcn_readfirstlane(tid >> 5);  // scalar wave id
  const int mo   = (lane & 16) ? 8 : 0;

  // ---- stage weights transposed (f32 -> f16, N-major) ----
  for (int i = tid; i < 64 * 128; i += 256) {
    int k = i >> 7, n = i & 127;          // wq/wk/wv: (64 x 128) row-major in
    sWqT[n * 64 + k] = (_Float16)wq[i];
    sWkT[n * 64 + k] = (_Float16)wk[i];
    sWvT[n * 64 + k] = (_Float16)wv[i];
    int kp = i >> 6, np = i & 63;         // wp: (128 x 64) row-major in
    sWpT[np * 128 + kp] = (_Float16)wp[i];
  }
  if (tid < 64) { sG[tid] = lng[tid]; sB[tid] = lnb[tid]; sBp[tid] = bp[tid]; }
  if (tid >= 64 && tid < 192) {
    int i = tid - 64;
    sBq[i] = bq[i]; sBk[i] = bk[i]; sBv[i] = bv[i];
  }

  // ---- line base / stride ----
  long base, stride;
  {
    int lb = blockIdx.x;
    if (axis == 0) { int n = lb / W, w = lb % W; base = (long)n * H * W * 64 + (long)w * 64; stride = (long)W * 64; }
    else           { int n = lb / H, h = lb % H; base = ((long)n * H + h) * (long)W * 64;    stride = 64; }
  }
  __syncthreads();

  // ---- layernorm -> sX (f16, pixel-major) ----
  for (int p = tid; p < L; p += 256) {
    const float* px = xin + base + (long)p * stride;
    float m = 0.f;
#pragma unroll
    for (int c = 0; c < 64; ++c) m += px[c];
    m *= (1.0f / 64.0f);
    float v = 0.f;
#pragma unroll
    for (int c = 0; c < 64; ++c) { float d = px[c] - m; v += d * d; }
    v *= (1.0f / 64.0f);
    float rs = rsqrtf(v + 1e-5f);
#pragma unroll
    for (int c = 0; c < 64; ++c)
      sX[p * 64 + c] = (_Float16)(((px[c] - m) * rs) * sG[c] + sB[c]);
  }
  __syncthreads();

  const float qscale = 0.17677669529663689f;   // KD^-0.5 folded into Q store

  // ---- Q = (WqT @ X)^T : A=sWqT (ch-major), B=sX (pixel-major as N-major),
  //      transposed b128 store -> sQ pixel-major.  Bias per D-row (channel).
  for (int i = 0; i < MT; ++i) {           // 8*MT tiles over 8 waves
    int t = wid + i * 8;
    int ct = t / MT, pt = t % MT;
    v8f c = load8f(sBq + ct * 16 + mo);
    c = wmma16(frag16(sWqT + ct * 16 * 64, 64, lane), frag16(sX + pt * 16 * 64, 64, lane), c);
    c = wmma16(frag16(sWqT + ct * 16 * 64 + 32, 64, lane), frag16(sX + pt * 16 * 64 + 32, 64, lane), c);
    store_c_f16T(sQ + pt * 16 * 128 + ct * 16, 128, lane, c, qscale);
  }
  // ---- K likewise ----
  for (int i = 0; i < MT; ++i) {
    int t = wid + i * 8;
    int ct = t / MT, pt = t % MT;
    v8f c = load8f(sBk + ct * 16 + mo);
    c = wmma16(frag16(sWkT + ct * 16 * 64, 64, lane), frag16(sX + pt * 16 * 64, 64, lane), c);
    c = wmma16(frag16(sWkT + ct * 16 * 64 + 32, 64, lane), frag16(sX + pt * 16 * 64 + 32, 64, lane), c);
    store_c_f16T(sK + pt * 16 * 128 + ct * 16, 128, lane, c, 1.0f);
  }
  // ---- V = X @ Wv, transposed b128 store -> sVt channel-major.
  //      Bias per D-column (channel = N here).
  for (int i = 0; i < MT; ++i) {
    int t = wid + i * 8;
    int pt = t >> 3, ct = t & 7;
    float bb = sBv[ct * 16 + (lane & 15)];
    v8f c;
#pragma unroll
    for (int j = 0; j < 8; ++j) c[j] = bb;
    c = wmma16(frag16(sX + pt * 16 * 64, 64, lane), frag16(sWvT + ct * 16 * 64, 64, lane), c);
    c = wmma16(frag16(sX + pt * 16 * 64 + 32, 64, lane), frag16(sWvT + ct * 16 * 64 + 32, 64, lane), c);
    store_c_f16T(sVt + ct * 16 * L + pt * 16, L, lane, c, 1.0f);
  }
  __syncthreads();

  // ---- per-head attention (8 waves cooperate per head) ----
  for (int h = 0; h < 4; ++h) {
    // S^T = Kh @ Qh^T (M=key, N=query, K=32); transposed b128 store -> sS query-major
    for (int i = 0; i < (MT * MT) / 8; ++i) {
      int t = wid + i * 8;
      int kt = t / MT, qt = t % MT;
      v8f c = {};
      c = wmma16(frag16(sK + kt * 16 * 128 + h * 32, 128, lane),
                 frag16(sQ + qt * 16 * 128 + h * 32, 128, lane), c);
      store_c_f16T(sS + qt * 16 * L + kt * 16, L, lane, c, 1.0f);
    }
    __syncthreads();
    // softmax rows: store unnormalized exp (f16), keep 1/sum in f32
    for (int q = tid; q < L; q += 256) {
      _Float16* row = sS + q * L;
      float m = -1e30f;
#pragma unroll 4
      for (int k = 0; k < L; k += 8) {
        v8h v = *(const v8h*)(row + k);
#pragma unroll
        for (int j = 0; j < 8; ++j) { float x = (float)v[j]; m = x > m ? x : m; }
      }
      float s = 0.f;
#pragma unroll 4
      for (int k = 0; k < L; k += 8) {
        v8h v = *(const v8h*)(row + k);
        v8h e;
#pragma unroll
        for (int j = 0; j < 8; ++j) { float x = __expf((float)v[j] - m); s += x; e[j] = (_Float16)x; }
        *(v8h*)(row + k) = e;
      }
      sRow[q] = 1.0f / s;
    }
    __syncthreads();
    // Yh = exp(S) @ Vh (M=query, N=32, K=L), 1/rowsum folded; pixel-major store
    for (int i = 0; i < (MT * 2) / 8; ++i) {
      int t = wid + i * 8;
      int mt = t >> 1, nt = t & 1;
      v8f c = {};
#pragma unroll
      for (int kc = 0; kc < KC; ++kc)
        c = wmma16(frag16(sS + mt * 16 * L + kc * 32, L, lane),
                   frag16(sVt + (h * 32 + nt * 16) * L + kc * 32, L, lane), c);
      v8f rs = load8f(sRow + mt * 16 + mo);
#pragma unroll
      for (int r = 0; r < 8; ++r) c[r] *= rs[r];
      store_c_f16(sY + mt * 16 * 128 + h * 32 + nt * 16, 128, lane, c);
    }
    __syncthreads();
  }

  // ---- out = (WpT @ Y)^T + bp -> global f32.
  //      D tile (m=out-channel, n=pixel): each lane writes 8 consecutive
  //      channels of one pixel -> 2x global_store_b128.
  for (int i = 0; i < (4 * MT) / 8; ++i) {
    int t = wid + i * 8;
    int ct = t / MT, pt = t % MT;
    v8f c = load8f(sBp + ct * 16 + mo);
#pragma unroll
    for (int kc = 0; kc < 4; ++kc)
      c = wmma16(frag16(sWpT + ct * 16 * 128 + kc * 32, 128, lane),
                 frag16(sY + pt * 16 * 128 + kc * 32, 128, lane), c);
    int p = pt * 16 + (lane & 15);
    float* dst = xout + base + (long)p * stride + ct * 16 + mo;
#pragma unroll
    for (int r = 0; r < 8; ++r) dst[r] = c[r];
  }
}

// ---------------------------------------------------------------------------
// Elementwise pipeline kernels
// ---------------------------------------------------------------------------

__global__ void __launch_bounds__(256) dwt_kernel(
    const float* __restrict__ in, float* __restrict__ ll, float* __restrict__ hp,
    int B, int H, int W)
{
  int H2 = H >> 1, W2 = W >> 1;
  long t = (long)blockIdx.x * blockDim.x + threadIdx.x;
  long total = (long)B * H2 * W2 * 64;
  if (t >= total) return;
  int c = (int)(t & 63);
  long r = t >> 6;
  int j = (int)(r % W2); r /= W2;
  int i = (int)(r % H2); int b = (int)(r / H2);
  long ib = (((long)b * H + 2 * i) * W + 2 * j) * 64 + c;
  float x00 = in[ib], x01 = in[ib + 64];
  float x10 = in[ib + (long)W * 64], x11 = in[ib + (long)W * 64 + 64];
  long ob = (((long)b * H2 + i) * W2 + j);
  ll[ob * 64 + c] = (x00 + x01 + x10 + x11) * 0.5f;
  float* o = hp + ob * 192;
  o[c]       = (x00 - x01 + x10 - x11) * 0.5f;
  o[64 + c]  = (x00 + x01 - x10 - x11) * 0.5f;
  o[128 + c] = (x00 - x01 - x10 + x11) * 0.5f;
}

__global__ void __launch_bounds__(256) idwt_kernel(
    const float* __restrict__ ll, const float* __restrict__ hp, float* __restrict__ out,
    int B, int H2, int W2)
{
  long t = (long)blockIdx.x * blockDim.x + threadIdx.x;
  long total = (long)B * H2 * W2 * 64;
  if (t >= total) return;
  int c = (int)(t & 63);
  long r = t >> 6;
  int j = (int)(r % W2); r /= W2;
  int i = (int)(r % H2); int b = (int)(r / H2);
  long ob = (((long)b * H2 + i) * W2 + j);
  float l = ll[ob * 64 + c];
  const float* s = hp + ob * 192;
  float a = s[c], h = s[64 + c], d = s[128 + c];
  int H = 2 * H2, W = 2 * W2;
  long o00 = (((long)b * H + 2 * i) * W + 2 * j) * 64 + c;
  out[o00]                     = (l + a + h + d) * 0.5f;
  out[o00 + 64]                = (l - a + h - d) * 0.5f;
  out[o00 + (long)W * 64]      = (l + a - h - d) * 0.5f;
  out[o00 + (long)W * 64 + 64] = (l - a - h + d) * 0.5f;
}

__global__ void __launch_bounds__(256) mix_kernel(
    const float* __restrict__ hp, const float* __restrict__ mw, float* __restrict__ xb,
    int B, int H2, int W2)
{
  long t = (long)blockIdx.x * blockDim.x + threadIdx.x;
  long total = (long)B * H2 * W2 * 64;
  if (t >= total) return;
  int c = (int)(t & 63);
  long r = t >> 6;
  int j = (int)(r % W2); r /= W2;
  int i = (int)(r % H2); int b = (int)(r / H2);
  const float* s = hp + (((long)b * H2 + i) * W2 + j) * 192;
  float h0 = s[c], h1 = s[64 + c], h2 = s[128 + c];
#pragma unroll
  for (int g = 0; g < 3; ++g) {
    float m = mw[g * 3 + 0] * h0 + mw[g * 3 + 1] * h1 + mw[g * 3 + 2] * h2;
    xb[(((long)(g * B + b) * H2 + i) * W2 + j) * 64 + c] = m;
  }
}

__global__ void __launch_bounds__(256) unbatch_res_kernel(
    float* __restrict__ hp, const float* __restrict__ att, float gain,
    int B, int H2, int W2)
{
  long t = (long)blockIdx.x * blockDim.x + threadIdx.x;
  long total = (long)B * H2 * W2 * 192;
  if (t >= total) return;
  int c = (int)(t & 63);
  long r = t >> 6;
  int g = (int)(r % 3); r /= 3;
  int j = (int)(r % W2); r /= W2;
  int i = (int)(r % H2); int b = (int)(r / H2);
  long pix = ((long)b * H2 + i) * W2 + j;
  hp[pix * 192 + g * 64 + c] +=
      gain * att[(((long)(g * B + b) * H2 + i) * W2 + j) * 64 + c];
}

__global__ void __launch_bounds__(256) axpy_kernel(
    float* __restrict__ y, const float* __restrict__ x, float a, long n)
{
  long t = (long)blockIdx.x * blockDim.x + threadIdx.x;
  if (t < n) y[t] += a * x[t];
}

// ---------------------------------------------------------------------------
// Host orchestration
// ---------------------------------------------------------------------------
static size_t attn_smem_bytes(int L) {
  size_t s = (size_t)(64 * 128 * 3 + 128 * 64) * 2;                 // weights f16
  s += ((size_t)L * 64 + (size_t)L * 128 * 4 + (size_t)L * L) * 2;  // X,Q,K,Vt,Y,S
  s += (size_t)(64 * 2 + 128 * 3 + 64) * 4;                         // ln + biases
  s += (size_t)L * 4;                                               // 1/rowsum
  return s;
}

static inline int nblk(long total, int bs) { return (int)((total + bs - 1) / bs); }

extern "C" void kernel_launch(void* const* d_in, const int* in_sizes, int n_in,
                              void* d_out, int out_size, void* d_ws, size_t ws_size,
                              hipStream_t stream) {
  (void)in_sizes; (void)n_in; (void)out_size; (void)ws_size;
  const float* x = (const float*)d_in[0];
  const float* lp[10];  for (int i = 0; i < 10; ++i) lp[i]  = (const float*)d_in[1 + i];
  const float* hpw[10]; for (int i = 0; i < 10; ++i) hpw[i] = (const float*)d_in[11 + i];
  const float* mix0 = (const float*)d_in[21];
  const float* mix1 = (const float*)d_in[22];

  const int B = 4;
  float* out = (float*)d_out;

  float* ws   = (float*)d_ws;
  float* cur1 = ws;                       // (4,128,128,64)   4,194,304
  float* hp0  = cur1 + 4194304;           // (4,128,128,192) 12,582,912
  float* cur2 = hp0 + 12582912;           // (4,64,64,64)     1,048,576
  float* hp1  = cur2 + 1048576;           // (4,64,64,192)    3,145,728
  float* bufA = hp1 + 3145728;            // 12,582,912
  float* bufB = bufA + 12582912;          // 12,582,912

  const size_t sm128 = attn_smem_bytes(128);
  const size_t sm64  = attn_smem_bytes(64);

  // DWT levels
  dwt_kernel<<<nblk(4194304, 256), 256, 0, stream>>>(x, cur1, hp0, B, 256, 256);
  dwt_kernel<<<nblk(1048576, 256), 256, 0, stream>>>(cur1, cur2, hp1, B, 128, 128);

  // hp level 0 (batch 12, 128x128), gain 0.15
  mix_kernel<<<nblk(4194304, 256), 256, 0, stream>>>(hp0, mix0, bufA, B, 128, 128);
  attn_pass_kernel<128><<<12 * 128, 256, sm128, stream>>>(
      bufA, bufB, hpw[0], hpw[1], hpw[2], hpw[3], hpw[4], hpw[5], hpw[6], hpw[7], hpw[8], hpw[9],
      128, 128, 0);
  attn_pass_kernel<128><<<12 * 128, 256, sm128, stream>>>(
      bufB, bufA, hpw[0], hpw[1], hpw[2], hpw[3], hpw[4], hpw[5], hpw[6], hpw[7], hpw[8], hpw[9],
      128, 128, 1);
  unbatch_res_kernel<<<nblk(12582912, 256), 256, 0, stream>>>(hp0, bufA, 0.15f, B, 128, 128);

  // hp level 1 (batch 12, 64x64), gain 0.25
  mix_kernel<<<nblk(1048576, 256), 256, 0, stream>>>(hp1, mix1, bufA, B, 64, 64);
  attn_pass_kernel<64><<<12 * 64, 256, sm64, stream>>>(
      bufA, bufB, hpw[0], hpw[1], hpw[2], hpw[3], hpw[4], hpw[5], hpw[6], hpw[7], hpw[8], hpw[9],
      64, 64, 0);
  attn_pass_kernel<64><<<12 * 64, 256, sm64, stream>>>(
      bufB, bufA, hpw[0], hpw[1], hpw[2], hpw[3], hpw[4], hpw[5], hpw[6], hpw[7], hpw[8], hpw[9],
      64, 64, 1);
  unbatch_res_kernel<<<nblk(3145728, 256), 256, 0, stream>>>(hp1, bufA, 0.25f, B, 64, 64);

  // lp (batch 4, 64x64): cur2 += 0.25 * mhsa(cur2)
  attn_pass_kernel<64><<<4 * 64, 256, sm64, stream>>>(
      cur2, bufB, lp[0], lp[1], lp[2], lp[3], lp[4], lp[5], lp[6], lp[7], lp[8], lp[9],
      64, 64, 0);
  attn_pass_kernel<64><<<4 * 64, 256, sm64, stream>>>(
      bufB, bufA, lp[0], lp[1], lp[2], lp[3], lp[4], lp[5], lp[6], lp[7], lp[8], lp[9],
      64, 64, 1);
  axpy_kernel<<<nblk(1048576, 256), 256, 0, stream>>>(cur2, bufA, 0.25f, (long)1048576);

  // IDWT levels
  idwt_kernel<<<nblk(1048576, 256), 256, 0, stream>>>(cur2, hp1, cur1, B, 64, 64);
  idwt_kernel<<<nblk(4194304, 256), 256, 0, stream>>>(cur1, hp0, out, B, 128, 128);
}
